// RWKV6TimeMix_34497177322192
// MI455X (gfx1250) — matrix operand order repaired and checked by reading, not compiled, not run on hardware
//
#include <hip/hip_runtime.h>
#include <hip/hip_bf16.h>

// CDNA5 / gfx1250 wave32 WMMA types
typedef __attribute__((ext_vector_type(16))) __bf16 v16bf;
typedef __attribute__((ext_vector_type(8)))  float  v8f;

// ------------------------------------------------------------------
// CDNA5 async global->LDS DMA (ASYNCcnt-tracked, no VGPR round trip)
// ------------------------------------------------------------------
__device__ __forceinline__ void async_load_b128(unsigned lds_addr,
                                                unsigned long long gaddr) {
    // LDS dest address from the lds_addr VGPR; global addr from VGPR pair, saddr off
    asm volatile("global_load_async_to_lds_b128 %0, %1, off"
                 :: "v"(lds_addr), "v"(gaddr)
                 : "memory");
}

template <int N>
__device__ __forceinline__ void wait_asynccnt() {
#if __has_builtin(__builtin_amdgcn_s_wait_asynccnt)
    __builtin_amdgcn_s_wait_asynccnt((short)N);
#else
    asm volatile("s_wait_asynccnt %0" :: "n"(N) : "memory");
#endif
}

// ------------------------------------------------------------------
// fp32 -> bf16 conversion (vectorized: float4 in, 4x bf16 out)
// ------------------------------------------------------------------
__global__ __launch_bounds__(256)
void cvt_f32_bf16(const float* __restrict__ in, __bf16* __restrict__ out, long n4) {
    long i = (long)blockIdx.x * blockDim.x + threadIdx.x;
    if (i >= n4) return;
    float4 v = ((const float4*)in)[i];
    union { __bf16 h[4]; uint2 u; } p;
    p.h[0] = (__bf16)v.x; p.h[1] = (__bf16)v.y;
    p.h[2] = (__bf16)v.z; p.h[3] = (__bf16)v.w;
    ((uint2*)out)[i] = p.u;
}

// ------------------------------------------------------------------
// GEMM: out[M,N] = A[M,K] * W[N,K]^T   (A, W bf16 row-major, out fp32)
// Block tile 128x128, BK=32, 256 threads = 8 waves (wave32).
// Wave grid 4(M) x 2(N): each wave computes 32x64 = 2x4 WMMA tiles.
// Double-buffered LDS fed by global_load_async_to_lds_b128.
// ------------------------------------------------------------------
#define BM 128
#define BN 128
#define BK 32

__global__ __launch_bounds__(256)
void gemm_bf16_wmma(const __bf16* __restrict__ A,
                    const __bf16* __restrict__ W,
                    float* __restrict__ out,
                    int M, int N, int K) {
    __shared__ __align__(16) __bf16 sA[2][BM * BK];   // 2 x 8 KB
    __shared__ __align__(16) __bf16 sB[2][BN * BK];   // 2 x 8 KB

    const int tid   = threadIdx.x;
    const int lane  = tid & 31;
    const int wave  = tid >> 5;
    const int waveM = (wave & 3) * 32;   // 4 waves along M
    const int waveN = (wave >> 2) * 64;  // 2 waves along N

    const int blockM = blockIdx.y * BM;
    const int blockN = blockIdx.x * BN;

    const int lcol  = lane & 15;
    const int lhalf = lane >> 4;

    // staging assignment: this thread's two 16B chunks of each tile
    const int idx0 = tid;          // chunk 0..255
    const int idx1 = tid + 256;    // chunk 256..511
    const int rowA0 = idx0 >> 2, c80 = (idx0 & 3) * 8;
    const int rowA1 = idx1 >> 2, c81 = (idx1 & 3) * 8;

    v8f acc[2][4] = {};

    const __bf16* gAbase = A + (size_t)blockM * K;
    const __bf16* gBbase = W + (size_t)blockN * K;

    // issue one tile's async DMA (4 x b128 per thread = 16 KB per workgroup)
    auto issue = [&](int buf, int k0) {
        async_load_b128((unsigned)(uintptr_t)((uint4*)sA[buf] + idx0),
                        (unsigned long long)(uintptr_t)(gAbase + (size_t)rowA0 * K + k0 + c80));
        async_load_b128((unsigned)(uintptr_t)((uint4*)sA[buf] + idx1),
                        (unsigned long long)(uintptr_t)(gAbase + (size_t)rowA1 * K + k0 + c81));
        async_load_b128((unsigned)(uintptr_t)((uint4*)sB[buf] + idx0),
                        (unsigned long long)(uintptr_t)(gBbase + (size_t)rowA0 * K + k0 + c80));
        async_load_b128((unsigned)(uintptr_t)((uint4*)sB[buf] + idx1),
                        (unsigned long long)(uintptr_t)(gBbase + (size_t)rowA1 * K + k0 + c81));
    };

    const int ntiles = K / BK;
    issue(0, 0);

    for (int it = 0; it < ntiles; ++it) {
        const int cur = it & 1;

        if (it + 1 < ntiles) {
            issue(cur ^ 1, (it + 1) * BK);   // overlap next tile's DMA with compute
            wait_asynccnt<4>();              // tile `it` landed; next 4 still in flight
        } else {
            wait_asynccnt<0>();
        }
        __syncthreads();                     // everyone's waits done -> tile visible

        union Frag { uint4 q[2]; v16bf v; };
        Frag fa[2], fb[4];

        // A 16x32 bf16: lane<16 -> M=lane,   K {0..7, 16..23}
        //               lane>=16 -> M=lane-16, K {8..15, 24..31}
        #pragma unroll
        for (int i = 0; i < 2; ++i) {
            const char* rowp = (const char*)(sA[cur] + (size_t)(waveM + i * 16 + lcol) * BK);
            fa[i].q[0] = *(const uint4*)(rowp + lhalf * 16);
            fa[i].q[1] = *(const uint4*)(rowp + 32 + lhalf * 16);
        }
        // B 32x16 bf16: lane<16 -> N=lane, K {0..7, 16..23} ; lane>=16 -> N=lane-16, K {8..15, 24..31}
        #pragma unroll
        for (int j = 0; j < 4; ++j) {
            const char* rowp = (const char*)(sB[cur] + (size_t)(waveN + j * 16 + lcol) * BK);
            fb[j].q[0] = *(const uint4*)(rowp + lhalf * 16);
            fb[j].q[1] = *(const uint4*)(rowp + 32 + lhalf * 16);
        }

        #pragma unroll
        for (int i = 0; i < 2; ++i)
            #pragma unroll
            for (int j = 0; j < 4; ++j)
                acc[i][j] = __builtin_amdgcn_wmma_f32_16x16x32_bf16(
                    false, fa[i].v, false, fb[j].v,
                    (short)0, acc[i][j], false, false);

        __syncthreads();   // all waves done reading buf `cur` before it is re-issued
    }

    // C/D layout: VGPR p -> row = 8*(lane/16) + p, col = lane%16
    #pragma unroll
    for (int i = 0; i < 2; ++i)
        #pragma unroll
        for (int j = 0; j < 4; ++j) {
            const int r0 = blockM + waveM + i * 16 + lhalf * 8;
            const int c  = blockN + waveN + j * 16 + lcol;
            #pragma unroll
            for (int p = 0; p < 8; ++p)
                out[(size_t)(r0 + p) * N + c] = acc[i][j][p];
        }
}

// ------------------------------------------------------------------
// WKV scan: one thread per (b, h*S+s) channel, sequential over T.
// Fuses sigmoid(r) * wkv and the bf16 down-convert for the last GEMM.
// ------------------------------------------------------------------
__global__ __launch_bounds__(256)
void wkv_scan(const float* __restrict__ kf, const float* __restrict__ vf,
              const float* __restrict__ rf,
              const float* __restrict__ time_decay,
              const float* __restrict__ time_first,
              __bf16* __restrict__ aout,
              int Bb, int T, int C) {
    int ch = blockIdx.x * blockDim.x + threadIdx.x;   // 0 .. Bb*C-1
    if (ch >= Bb * C) return;
    int b = ch / C;
    int c = ch % C;   // == h*S + s ; time_decay/first flatten [H,S] -> C

    float decay = __expf(-__expf(time_decay[c]));
    float first = __expf(time_first[c]);

    float num = 0.f, den = 0.f;
    size_t base = (size_t)b * T * C + c;

    for (int t = 0; t < T; ++t) {
        size_t idx = base + (size_t)t * C;
        float kk = kf[idx];
        kk = fminf(fmaxf(kk, -10.f), 10.f);
        float w = __expf(kk);
        if (t == 0) w *= first;
        float vv = vf[idx];
        num = decay * num + w * vv;
        den = decay * den + w;
        float wkv = num / (den + 1e-6f);
        float rr  = rf[idx];
        float sig = 1.f / (1.f + __expf(-rr));
        aout[idx] = (__bf16)(sig * wkv);
    }
}

// ------------------------------------------------------------------
// launcher
// ------------------------------------------------------------------
extern "C" void kernel_launch(void* const* d_in, const int* in_sizes, int n_in,
                              void* d_out, int out_size, void* d_ws, size_t ws_size,
                              hipStream_t stream) {
    const float* x  = (const float*)d_in[0];
    const float* Wk = (const float*)d_in[1];
    const float* Wv = (const float*)d_in[2];
    const float* Wr = (const float*)d_in[3];
    const float* Wo = (const float*)d_in[4];
    const float* td = (const float*)d_in[5];
    const float* tf = (const float*)d_in[6];

    const int C = 2048, T = 2048;
    const long xN = in_sizes[0];                 // B*T*C
    const int Bb  = (int)(xN / ((long)T * C));   // 4
    const int M   = Bb * T;                      // 8192

    // workspace layout
    char* ws = (char*)d_ws;
    __bf16* xb  = (__bf16*)ws; ws += (size_t)M * C * 2;
    __bf16* Wkb = (__bf16*)ws; ws += (size_t)C * C * 2;
    __bf16* Wvb = (__bf16*)ws; ws += (size_t)C * C * 2;
    __bf16* Wrb = (__bf16*)ws; ws += (size_t)C * C * 2;
    __bf16* Wob = (__bf16*)ws; ws += (size_t)C * C * 2;
    float*  kfp = (float*)ws;  ws += (size_t)M * C * 4;
    float*  vfp = (float*)ws;  ws += (size_t)M * C * 4;
    float*  rfp = (float*)ws;  ws += (size_t)M * C * 4;
    __bf16* ab  = (__bf16*)ws; ws += (size_t)M * C * 2;

    // 1. down-convert activations and weights to bf16
    {
        long n4 = (long)M * C / 4;
        cvt_f32_bf16<<<(int)((n4 + 255) / 256), 256, 0, stream>>>(x, xb, n4);
        long w4 = (long)C * C / 4;
        int  wb = (int)((w4 + 255) / 256);
        cvt_f32_bf16<<<wb, 256, 0, stream>>>(Wk, Wkb, w4);
        cvt_f32_bf16<<<wb, 256, 0, stream>>>(Wv, Wvb, w4);
        cvt_f32_bf16<<<wb, 256, 0, stream>>>(Wr, Wrb, w4);
        cvt_f32_bf16<<<wb, 256, 0, stream>>>(Wo, Wob, w4);
    }

    // 2. k / v / r projections (bf16 WMMA, fp32 accumulate)
    dim3 grid(C / BN, M / BM);
    gemm_bf16_wmma<<<grid, 256, 0, stream>>>(xb, Wkb, kfp, M, C, C);
    gemm_bf16_wmma<<<grid, 256, 0, stream>>>(xb, Wvb, vfp, M, C, C);
    gemm_bf16_wmma<<<grid, 256, 0, stream>>>(xb, Wrb, rfp, M, C, C);

    // 3. WKV recurrence + sigmoid(r) gating, emit bf16 operand for Wo GEMM
    {
        int nch = Bb * C;
        wkv_scan<<<(nch + 255) / 256, 256, 0, stream>>>(kfp, vfp, rfp, td, tf, ab, Bb, T, C);
    }

    // 4. output projection straight into d_out (fp32)
    gemm_bf16_wmma<<<grid, 256, 0, stream>>>(ab, Wob, (float*)d_out, M, C, C);
}
